// HGNNLayer_24060406792470
// MI455X (gfx1250) — matrix operand order repaired and checked by reading, not compiled, not run on hardware
//
#include <hip/hip_runtime.h>
#include <cstdint>
#include <cstddef>

#define N_ITEM 80000
#define N_USER 20000
#define NTOT   100000
#define DD     128
#define E_PER  1600000

typedef __attribute__((ext_vector_type(16))) __bf16 v16bf;
typedef __attribute__((ext_vector_type(8)))  float  v8f;

union Frag16 { v16bf v; unsigned int u[8]; };

__device__ __forceinline__ unsigned int f2bf_scalar(float f) {
    union { float f; unsigned int u; } c; c.f = f;
    unsigned int u = c.u;
    u += 0x7fffu + ((u >> 16) & 1u);      // round-to-nearest-even
    return u >> 16;
}

// packed float2 -> bf16x2 (one v_cvt_pk_bf16_f32 when the builtin exists)
__device__ __forceinline__ unsigned int pack_bf16(float x, float y) {
#if __has_builtin(__builtin_amdgcn_cvt_pk_bf16_f32)
    typedef __attribute__((ext_vector_type(2))) __bf16 v2bf;
    union { v2bf v; unsigned int u; } r;
    r.v = __builtin_amdgcn_cvt_pk_bf16_f32(x, y);
    return r.u;
#else
    return f2bf_scalar(x) | (f2bf_scalar(y) << 16);
#endif
}

#define LDSW 136   // padded bf16 row stride: 272B = 16B-aligned, 4-bank skew per row

// out[n,:] = act( (A0/A1[n,:] + Aadd[n,:]) @ W^T + bias )
// A0 rows [0,nrows0), A1 rows [nrows0,nrows) (concat); Aadd/bias optional.
__global__ __launch_bounds__(256) void hgnn_gemm128(
    const float* __restrict__ A0, const float* __restrict__ A1,
    const float* __restrict__ Aadd,
    const float* __restrict__ W, const float* __restrict__ bias,
    float* __restrict__ out, int nrows, int nrows0, int relu)
{
    __shared__ unsigned short Wlds[DD * LDSW];

    // cooperative fp32 -> bf16 convert of the 128x128 weight into LDS (2 cols/thread)
    for (int idx = threadIdx.x; idx < DD * DD / 2; idx += 256) {
        int r = idx >> 6, c = (idx & 63) * 2;
        float2 w = *(const float2*)(W + r * DD + c);
        *(unsigned int*)&Wlds[r * LDSW + c] = pack_bf16(w.x, w.y);
    }
    __syncthreads();

    const int lane   = threadIdx.x & 31;
    const int wave   = threadIdx.x >> 5;
    const int half   = lane >> 4;
    const int mrow   = lane & 15;
    const int rowTile = blockIdx.x * 128 + wave * 16;

    // ---- A fragments: this wave's 16 rows x K=128 as 4 k-tiles of 32 (bf16) ----
    int row  = rowTile + mrow;
    int rowc = row < nrows ? row : nrows - 1;     // clamp tail (stores are guarded)
    const float* aptr = (A1 && rowc >= nrows0) ? (A1 + (size_t)(rowc - nrows0) * DD)
                                               : (A0 + (size_t)rowc * DD);
    const float* addp = Aadd ? (Aadd + (size_t)rowc * DD) : nullptr;

    Frag16 afrag[4];
#pragma unroll
    for (int kt = 0; kt < 4; ++kt) {
#pragma unroll
        for (int j = 0; j < 8; ++j) {
            int k = 32 * kt + 2 * j + ((j >= 4) ? 8 : 0) + 8 * half;
            float2 av = *(const float2*)(aptr + k);
            if (addp) { float2 ad = *(const float2*)(addp + k); av.x += ad.x; av.y += ad.y; }
            afrag[kt].u[j] = pack_bf16(av.x, av.y);
        }
    }

    const bool fullTile = (rowTile + 16) <= nrows;   // wave-uniform

    // ---- 8 n-tiles of 16 columns; accumulate K=128 via 4 WMMAs each ----
#pragma unroll
    for (int nt = 0; nt < 8; ++nt) {
        v8f acc = {};
        const char* brow = (const char*)Wlds + (size_t)(16 * nt + mrow) * (LDSW * 2);
#pragma unroll
        for (int kt = 0; kt < 4; ++kt) {
            Frag16 bfrag;
            const int4* bp = (const int4*)(brow + kt * 64 + half * 16);
            int4 q0 = bp[0]; int4 q1 = bp[2];
            bfrag.u[0] = q0.x; bfrag.u[1] = q0.y; bfrag.u[2] = q0.z; bfrag.u[3] = q0.w;
            bfrag.u[4] = q1.x; bfrag.u[5] = q1.y; bfrag.u[6] = q1.z; bfrag.u[7] = q1.w;
            acc = __builtin_amdgcn_wmma_f32_16x16x32_bf16(
                      false, afrag[kt].v, false, bfrag.v, (short)0, acc, false, false);
        }
        int col = 16 * nt + mrow;
        float b = bias ? bias[col] : 0.0f;
        // D layout: VGPR j -> M = j + 8*half, N = mrow
        float* obase = out + (size_t)(rowTile + 8 * half) * DD + col;
        if (fullTile) {
#pragma unroll
            for (int j = 0; j < 8; ++j) {
                float v = acc[j] + b;
                if (relu) v = fmaxf(v, 0.0f);
                obase[(size_t)j * DD] = v;
            }
        } else {
#pragma unroll
            for (int j = 0; j < 8; ++j) {
                int ro = rowTile + 8 * half + j;
                if (ro < nrows) {
                    float v = acc[j] + b;
                    if (relu) v = fmaxf(v, 0.0f);
                    obase[(size_t)j * DD] = v;
                }
            }
        }
    }
}

// msg[rows[e],:] += vals[e] * hr[cols[e],:]   — one edge per wave32, float4 per lane
__global__ __launch_bounds__(256) void hgnn_scatter(
    const float* __restrict__ hr, const float* __restrict__ vals,
    const int* __restrict__ rows, const int* __restrict__ cols,
    float* __restrict__ msg, int nEdges)
{
    int gid  = blockIdx.x * blockDim.x + threadIdx.x;
    int e    = gid >> 5;
    int lane = gid & 31;
    if (e >= nEdges) return;

    if (e + 8 < nEdges) {                         // prefetch a future edge's source row
        int ps = cols[e + 8];
        __builtin_prefetch(hr + (size_t)ps * DD + lane * 4, 0, 0);
    }

    int   src = cols[e];
    int   dst = rows[e];
    float v   = vals[e];

    float4 x = *((const float4*)(hr + (size_t)src * DD) + lane);
    float* d = msg + (size_t)dst * DD + lane * 4;
    atomicAdd(d + 0, x.x * v);
    atomicAdd(d + 1, x.y * v);
    atomicAdd(d + 2, x.z * v);
    atomicAdd(d + 3, x.w * v);
}

__global__ __launch_bounds__(256) void hgnn_zero(float4* __restrict__ p, int n4)
{
    for (int i = blockIdx.x * blockDim.x + threadIdx.x; i < n4; i += gridDim.x * blockDim.x)
        p[i] = make_float4(0.f, 0.f, 0.f, 0.f);
}

extern "C" void kernel_launch(void* const* d_in, const int* in_sizes, int n_in,
                              void* d_out, int out_size, void* d_ws, size_t ws_size,
                              hipStream_t stream) {
    (void)in_sizes; (void)n_in; (void)out_size; (void)ws_size;

    const float* h_item = (const float*)d_in[0];   // [80000,128]
    const float* h_user = (const float*)d_in[1];   // [20000,128]
    const float* W_rel  = (const float*)d_in[2];   // [3,128,128]
    const float* W_item = (const float*)d_in[3];   // [128,128]
    const float* b_item = (const float*)d_in[4];   // [128]
    const float* W_user = (const float*)d_in[5];   // [128,128]
    const float* b_user = (const float*)d_in[6];   // [128]
    const float* vals   = (const float*)d_in[7];   // [3,E]
    const int*   rows   = (const int*)d_in[8];     // [3,E]
    const int*   cols   = (const int*)d_in[9];     // [3,E]

    float* out = (float*)d_out;                    // [100000,128] item||user

    // workspace: msg [N,128] fp32 (51.2 MB) + hr [N,128] fp32 (51.2 MB)
    float* msg = (float*)d_ws;
    float* hr  = msg + (size_t)NTOT * DD;

    // zero the message accumulator (re-done every call: deterministic)
    hgnn_zero<<<2048, 256, 0, stream>>>((float4*)msg, NTOT * DD / 4);

    const int gemmBlocksAll  = (NTOT   + 127) / 128;   // 782
    const int gemmBlocksItem = (N_ITEM + 127) / 128;   // 625
    const int gemmBlocksUser = (N_USER + 127) / 128;   // 157
    const int scatBlocks     = (E_PER * 32 + 255) / 256;

    for (int r = 0; r < 3; ++r) {
        // hr = concat(h_item,h_user) @ W_rel[r]^T
        hgnn_gemm128<<<gemmBlocksAll, 256, 0, stream>>>(
            h_item, h_user, nullptr,
            W_rel + (size_t)r * DD * DD, nullptr,
            hr, NTOT, N_ITEM, 0);
        // msg += segment_sum(vals[r] * hr[cols[r]], rows[r])
        hgnn_scatter<<<scatBlocks, 256, 0, stream>>>(
            hr, vals + (size_t)r * E_PER, rows + (size_t)r * E_PER,
            cols + (size_t)r * E_PER, msg, E_PER);
    }

    // h_item_out = relu((msg[:Ni] + h_item) @ W_item^T + b_item)
    hgnn_gemm128<<<gemmBlocksItem, 256, 0, stream>>>(
        h_item, nullptr, msg, W_item, b_item, out, N_ITEM, N_ITEM, 1);

    // h_user_out = relu((msg[Ni:] + h_user) @ W_user^T + b_user)
    hgnn_gemm128<<<gemmBlocksUser, 256, 0, stream>>>(
        h_user, nullptr, msg + (size_t)N_ITEM * DD, W_user, b_user,
        out + (size_t)N_ITEM * DD, N_USER, N_USER, 1);
}
